// GetModel_78658031058999
// MI455X (gfx1250) — matrix operand order repaired and checked by reading, not compiled, not run on hardware
//
#include <hip/hip_runtime.h>

typedef _Float16 h8v  __attribute__((ext_vector_type(8)));
typedef _Float16 h16v __attribute__((ext_vector_type(16)));
typedef float    f8v  __attribute__((ext_vector_type(8)));

__device__ inline h16v ld16(const _Float16* __restrict__ p) {
  h8v x = *(const h8v*)p;
  h8v y = *(const h8v*)(p + 8);
  h16v r;
  #pragma unroll
  for (int i = 0; i < 8; ++i) { r[i] = x[i]; r[i + 8] = y[i]; }
  return r;
}

// ---------------------------------------------------------------------------
// xyz (B,6,N) -> coords (B,N,3), norm (B,N,3)
__global__ __launch_bounds__(256)
void split_kernel(const float* __restrict__ xyz, float* __restrict__ coords,
                  float* __restrict__ nrm, int B, int N) {
  size_t gid = (size_t)blockIdx.x * blockDim.x + threadIdx.x;
  if (gid >= (size_t)B * N) return;
  int b = (int)(gid / N), n = (int)(gid % N);
  const float* src = xyz + (size_t)b * 6 * N;
  for (int d = 0; d < 3; ++d) {
    coords[gid * 3 + d] = src[(size_t)d * N + n];
    nrm[gid * 3 + d]    = src[(size_t)(3 + d) * N + n];
  }
}

// ---------------------------------------------------------------------------
// Farthest point sampling: one block (1024 threads) per batch.
__global__ __launch_bounds__(1024)
void fps_kernel(const float* __restrict__ pts, int N, int npoint,
                int* __restrict__ out) {
  const int b = blockIdx.x, t = threadIdx.x;
  const float* P = pts + (size_t)b * N * 3;
  float px[4], py[4], pz[4], dist[4];
  #pragma unroll
  for (int k = 0; k < 4; ++k) {
    int i = t + k * 1024;
    if (i < N) { px[k]=P[i*3]; py[k]=P[i*3+1]; pz[k]=P[i*3+2]; dist[k]=1e10f; }
    else       { px[k]=0.f; py[k]=0.f; pz[k]=0.f; dist[k]=-1.0f; }
  }
  __shared__ float rv[1024];
  __shared__ int   ri[1024];
  __shared__ float sfx, sfy, sfz;
  __shared__ int   sfar;
  if (t == 0) { sfar = 0; sfx = P[0]; sfy = P[1]; sfz = P[2]; }
  __syncthreads();
  for (int it = 0; it < npoint; ++it) {
    if (t == 0) out[b * npoint + it] = sfar;
    float fx = sfx, fy = sfy, fz = sfz;
    float bestv = -1e30f; int besti = 0x7fffffff;
    #pragma unroll
    for (int k = 0; k < 4; ++k) {
      int i = t + k * 1024;
      if (i < N) {
        float dx = px[k]-fx, dy = py[k]-fy, dz = pz[k]-fz;
        float d = dx*dx + dy*dy + dz*dz;
        dist[k] = fminf(dist[k], d);
        if (dist[k] > bestv) { bestv = dist[k]; besti = i; }
      }
    }
    rv[t] = bestv; ri[t] = besti;
    __syncthreads();
    for (int s = 512; s > 0; s >>= 1) {
      if (t < s) {
        float v2 = rv[t + s]; int i2 = ri[t + s];
        if (v2 > rv[t] || (v2 == rv[t] && i2 < ri[t])) { rv[t] = v2; ri[t] = i2; }
      }
      __syncthreads();
    }
    if (t == 0) sfar = ri[0];
    __syncthreads();
    int far = sfar;
    if ((far & 1023) == t) { int k = far >> 10; sfx = px[k]; sfy = py[k]; sfz = pz[k]; }
    __syncthreads();
  }
}

// new_xyz[b,s,:] = pts[b, idx[b,s], :]
__global__ __launch_bounds__(256)
void gather3_kernel(const float* __restrict__ pts, const int* __restrict__ fidx,
                    float* __restrict__ out, int N, int npoint, int B) {
  size_t gid = (size_t)blockIdx.x * blockDim.x + threadIdx.x;
  if (gid >= (size_t)B * npoint) return;
  int b = (int)(gid / npoint);
  int id = fidx[gid];
  const float* p = pts + ((size_t)b * N + id) * 3;
  out[gid*3] = p[0]; out[gid*3+1] = p[1]; out[gid*3+2] = p[2];
}

// ---------------------------------------------------------------------------
// Ball query: block per batch, points cached in LDS, in-order radius scan.
__global__ __launch_bounds__(512)
void ballquery_kernel(const float* __restrict__ xyz, const float* __restrict__ new_xyz,
                      int N, int npoint, int nsample, float r2, int* __restrict__ idx) {
  extern __shared__ float spts[];
  int b = blockIdx.x;
  for (int i = threadIdx.x; i < N * 3; i += blockDim.x)
    spts[i] = xyz[(size_t)b * N * 3 + i];
  __syncthreads();
  int s = threadIdx.x;
  if (s >= npoint) return;
  const float* c = new_xyz + ((size_t)b * npoint + s) * 3;
  float cx = c[0], cy = c[1], cz = c[2];
  int* o = idx + ((size_t)b * npoint + s) * nsample;
  int cnt = 0;
  for (int j = 0; j < N && cnt < nsample; ++j) {
    float dx = spts[j*3]-cx, dy = spts[j*3+1]-cy, dz = spts[j*3+2]-cz;
    if (dx*dx + dy*dy + dz*dz <= r2) o[cnt++] = j;
  }
  int first = (cnt > 0) ? o[0] : 0;
  for (; cnt < nsample; ++cnt) o[cnt] = first;
}

// ---------------------------------------------------------------------------
// Build grouped f16 tensor: [centered xyz(3) | feats(F) | zero pad] per row.
__global__ __launch_bounds__(256)
void group_kernel(const float* __restrict__ xyz, const float* __restrict__ feats,
                  const float* __restrict__ new_xyz, const int* __restrict__ idx,
                  _Float16* __restrict__ out, int N, int npoint, int nsample,
                  int F, int Kpad, int B) {
  size_t gid = (size_t)blockIdx.x * blockDim.x + threadIdx.x;
  size_t total = (size_t)B * npoint * nsample;
  if (gid >= total) return;
  int s = (int)((gid / nsample) % npoint);
  int b = (int)(gid / ((size_t)nsample * npoint));
  int id = idx[gid];
  const float* p = xyz + ((size_t)b * N + id) * 3;
  const float* c = new_xyz + ((size_t)b * npoint + s) * 3;
  _Float16* o = out + gid * Kpad;
  o[0] = (_Float16)(p[0] - c[0]);
  o[1] = (_Float16)(p[1] - c[1]);
  o[2] = (_Float16)(p[2] - c[2]);
  const float* f = feats + ((size_t)b * N + id) * F;
  for (int k = 0; k < F; ++k) o[3 + k] = (_Float16)f[k];
  for (int k = 3 + F; k < Kpad; ++k) o[k] = (_Float16)0.f;
}

// sa3 group-all: [xyz(3) | feats(F) | pad], one row per (b, i)
__global__ __launch_bounds__(256)
void group_all_kernel(const float* __restrict__ xyz, const float* __restrict__ feats,
                      _Float16* __restrict__ out, int S, int F, int Kpad, int B) {
  size_t gid = (size_t)blockIdx.x * blockDim.x + threadIdx.x;
  if (gid >= (size_t)B * S) return;
  _Float16* o = out + gid * Kpad;
  const float* c = xyz + gid * 3;
  o[0] = (_Float16)c[0]; o[1] = (_Float16)c[1]; o[2] = (_Float16)c[2];
  const float* f = feats + gid * F;
  for (int k = 0; k < F; ++k) o[3 + k] = (_Float16)f[k];
  for (int k = 3 + F; k < Kpad; ++k) o[k] = (_Float16)0.f;
}

// ---------------------------------------------------------------------------
// Weight f32 (K,N) -> f16 swizzled to WMMA B-fragment order:
// out[((ntile*(Kp/32) + kblk)*32 + lane)*16 + i]  where
//   k = kblk*32 + (lane>>4)*16 + i, col = ntile*16 + (lane&15)
// so each lane's 16-half fragment is contiguous (2x b128 loads in the GEMM).
__global__ __launch_bounds__(256)
void wcvt_swz_kernel(const float* __restrict__ w, _Float16* __restrict__ out,
                     int K, int Kp, int Ncol) {
  size_t gid = (size_t)blockIdx.x * blockDim.x + threadIdx.x;
  if (gid >= (size_t)Kp * Ncol) return;
  int nkb = Kp >> 5;
  int i    = (int)(gid & 15);
  int l    = (int)((gid >> 4) & 31);
  size_t rest = gid >> 9;
  int kblk = (int)(rest % nkb);
  int nt   = (int)(rest / nkb);
  int k    = kblk * 32 + (l >> 4) * 16 + i;
  int col  = nt * 16 + (l & 15);
  out[gid] = (k < K) ? (_Float16)w[(size_t)k * Ncol + col] : (_Float16)0.f;
}

// ---------------------------------------------------------------------------
// WMMA GEMM: D = relu((A@W + bias) * gamma/sqrt(1+eps) + beta).
// 4 waves/block; each wave computes a 16x64 strip: one A fragment reused
// across 4 B tiles (cuts A HBM traffic 4x vs 16-wide tiles).
// W pre-swizzled to fragment order. M%64==0, N%64==0, K%32==0.
__global__ __launch_bounds__(128)
void gemm_bn_relu_wmma(const _Float16* __restrict__ A, const _Float16* __restrict__ Wsw,
                       const float* __restrict__ bias, const float* __restrict__ gamma,
                       const float* __restrict__ beta, _Float16* __restrict__ D,
                       int M, int K, int N) {
  const int lane = threadIdx.x & 31;
  const int wv   = threadIdx.x >> 5;
  const int m0 = (blockIdx.x * 4 + wv) * 16;
  const int n0 = blockIdx.y * 64;
  const int lrow = lane & 15;
  const int lhi  = lane >> 4;
  const int nkb  = K >> 5;

  const _Float16* Arow  = A + (size_t)(m0 + lrow) * K + lhi * 16;
  // fragment base for n-tile (n0/16): ((ntile*nkb + kblk)*32 + lane)*16
  const _Float16* Bbase = Wsw + ((size_t)(n0 >> 4) * nkb * 32 + lane) * 16;
  const size_t tileStride = (size_t)nkb * 512;   // halves between n-tiles

  f8v acc[4];
  #pragma unroll
  for (int j = 0; j < 4; ++j) acc[j] = f8v{0.f,0.f,0.f,0.f,0.f,0.f,0.f,0.f};

  for (int kb = 0; kb < nkb; ++kb) {
    h8v a0 = *(const h8v*)(Arow + kb * 32);
    h8v a1 = *(const h8v*)(Arow + kb * 32 + 8);
    h16v a;
    #pragma unroll
    for (int i = 0; i < 8; ++i) { a[i] = a0[i]; a[i + 8] = a1[i]; }
    const _Float16* bp = Bbase + (size_t)kb * 512;
    #pragma unroll
    for (int j = 0; j < 4; ++j) {
      h16v b = ld16(bp + (size_t)j * tileStride);
      acc[j] = __builtin_amdgcn_wmma_f32_16x16x32_f16(false, a, false, b,
                                                      (short)0, acc[j], false, false);
    }
  }

  const float rs = rsqrtf(1.0f + 1e-5f);
  #pragma unroll
  for (int j = 0; j < 4; ++j) {
    const int col = n0 + j * 16 + lrow;
    const float s  = gamma[col] * rs;
    const float bb = bias[col];
    const float bt = beta[col];
    #pragma unroll
    for (int r = 0; r < 8; ++r) {
      int row = m0 + r + lhi * 8;
      float v = (acc[j][r] + bb) * s + bt;
      v = fmaxf(v, 0.0f);
      D[(size_t)row * N + col] = (_Float16)v;
    }
  }
}

// ---------------------------------------------------------------------------
// Max over nsample: act (rows, ns, C) f16 -> out (rows, C) f32 (opt. 2nd copy)
__global__ __launch_bounds__(256)
void maxpool_kernel(const _Float16* __restrict__ act, float* __restrict__ out,
                    float* __restrict__ out2, int rows, int ns, int C) {
  size_t gid = (size_t)blockIdx.x * blockDim.x + threadIdx.x;
  if (gid >= (size_t)rows * C) return;
  size_t row = gid / C; int c = (int)(gid % C);
  const _Float16* a = act + (row * ns) * (size_t)C + c;
  float m = -3.4e38f;
  for (int j = 0; j < ns; ++j) m = fmaxf(m, (float)a[(size_t)j * C]);
  out[gid] = m;
  if (out2) out2[gid] = m;
}

// fc: out = mode==0 ? bn(relu(x@w+b)) : x@w+b   (f32)
__global__ __launch_bounds__(256)
void fc_kernel(const float* __restrict__ x, const float* __restrict__ w,
               const float* __restrict__ bias, const float* __restrict__ g,
               const float* __restrict__ beta, float* __restrict__ out,
               int M, int K, int N, int mode) {
  size_t gid = (size_t)blockIdx.x * blockDim.x + threadIdx.x;
  if (gid >= (size_t)M * N) return;
  int r = (int)(gid / N), c = (int)(gid % N);
  float acc = bias[c];
  const float* xr = x + (size_t)r * K;
  for (int k = 0; k < K; ++k) acc += xr[k] * w[(size_t)k * N + c];
  if (mode == 0) {
    acc = fmaxf(acc, 0.0f);
    acc = g[c] * acc * rsqrtf(1.0f + 1e-5f) + beta[c];
  }
  out[gid] = acc;
}

__global__ void logsoftmax_kernel(const float* __restrict__ in, float* __restrict__ out,
                                  int M, int N) {
  int r = blockIdx.x * blockDim.x + threadIdx.x;
  if (r >= M) return;
  const float* x = in + (size_t)r * N;
  float mx = -3.4e38f;
  for (int n = 0; n < N; ++n) mx = fmaxf(mx, x[n]);
  float s = 0.f;
  for (int n = 0; n < N; ++n) s += expf(x[n] - mx);
  float ls = logf(s);
  for (int n = 0; n < N; ++n) out[(size_t)r * N + n] = x[n] - mx - ls;
}

// ---------------------------------------------------------------------------
static inline dim3 grid1d(size_t total, int block) {
  return dim3((unsigned)((total + block - 1) / block));
}

extern "C" void kernel_launch(void* const* d_in, const int* in_sizes, int n_in,
                              void* d_out, int out_size, void* d_ws, size_t ws_size,
                              hipStream_t stream) {
  (void)in_sizes; (void)n_in; (void)out_size; (void)ws_size;
  const int B = 32, N = 4096;

  const float* xyz = (const float*)d_in[0];
  const float *Wc[9], *Bc[9], *Gc[9], *Ec[9];
  int pi = 1;
  for (int l = 0; l < 9; ++l) {
    Wc[l] = (const float*)d_in[pi]; Bc[l] = (const float*)d_in[pi+1];
    Gc[l] = (const float*)d_in[pi+2]; Ec[l] = (const float*)d_in[pi+3]; pi += 4;
  }
  const float* fw1=(const float*)d_in[37]; const float* fb1=(const float*)d_in[38];
  const float* fg1=(const float*)d_in[39]; const float* fe1=(const float*)d_in[40];
  const float* fw2=(const float*)d_in[41]; const float* fb2=(const float*)d_in[42];
  const float* fg2=(const float*)d_in[43]; const float* fe2=(const float*)d_in[44];
  const float* fw3=(const float*)d_in[45]; const float* fb3=(const float*)d_in[46];

  static const int Kin[9] = {6,64,64, 131,128,128, 259,256,512};
  static const int Kp [9] = {32,64,64, 160,128,128, 288,256,512};
  static const int Nc [9] = {64,64,128, 128,128,256, 256,512,1024};

  char* base = (char*)d_ws; size_t off = 0;
  auto alloc = [&](size_t bytes) -> void* {
    void* p = base + off; off += (bytes + 255) & ~(size_t)255; return p;
  };
  float* coords = (float*)alloc((size_t)B*N*3*4);
  float* nrm    = (float*)alloc((size_t)B*N*3*4);
  int*   fps1   = (int*)  alloc((size_t)B*512*4);
  float* nx1    = (float*)alloc((size_t)B*512*3*4);
  int*   idx1   = (int*)  alloc((size_t)B*512*32*4);
  float* l1pts  = (float*)alloc((size_t)B*512*128*4);
  int*   fps2   = (int*)  alloc((size_t)B*128*4);
  float* nx2    = (float*)alloc((size_t)B*128*3*4);
  int*   idx2   = (int*)  alloc((size_t)B*128*64*4);
  float* l2pts  = (float*)alloc((size_t)B*128*256*4);
  float* l3     = (float*)alloc((size_t)B*1024*4);
  float* fcb1   = (float*)alloc((size_t)B*512*4);
  float* fcb2   = (float*)alloc((size_t)B*256*4);
  float* fcb3   = (float*)alloc((size_t)B*40*4);
  _Float16* wh[9];
  for (int l = 0; l < 9; ++l) wh[l] = (_Float16*)alloc((size_t)Kp[l]*Nc[l]*2);
  _Float16* Gb = (_Float16*)alloc((size_t)262144*160*2);
  _Float16* A1 = (_Float16*)alloc((size_t)33554432*2);
  _Float16* A2 = (_Float16*)alloc((size_t)33554432*2);
  _Float16* A3 = (_Float16*)alloc((size_t)67108864*2);

  for (int l = 0; l < 9; ++l) {
    size_t tot = (size_t)Kp[l] * Nc[l];
    wcvt_swz_kernel<<<grid1d(tot,256),256,0,stream>>>(Wc[l], wh[l], Kin[l], Kp[l], Nc[l]);
  }

  split_kernel<<<grid1d((size_t)B*N,256),256,0,stream>>>(xyz, coords, nrm, B, N);

  // ---- SA1: npoint=512, r=0.2, ns=32
  fps_kernel<<<B,1024,0,stream>>>(coords, N, 512, fps1);
  gather3_kernel<<<grid1d((size_t)B*512,256),256,0,stream>>>(coords, fps1, nx1, N, 512, B);
  ballquery_kernel<<<B,512,(size_t)N*3*4,stream>>>(coords, nx1, N, 512, 32, 0.04f, idx1);
  {
    size_t M1 = (size_t)B*512*32;
    group_kernel<<<grid1d(M1,256),256,0,stream>>>(coords, nrm, nx1, idx1, Gb,
                                                  N, 512, 32, 3, 32, B);
    gemm_bn_relu_wmma<<<dim3((unsigned)(M1/64),Nc[0]/64),128,0,stream>>>(
        Gb, wh[0], Bc[0], Gc[0], Ec[0], A1, (int)M1, Kp[0], Nc[0]);
    gemm_bn_relu_wmma<<<dim3((unsigned)(M1/64),Nc[1]/64),128,0,stream>>>(
        A1, wh[1], Bc[1], Gc[1], Ec[1], A2, (int)M1, Kp[1], Nc[1]);
    gemm_bn_relu_wmma<<<dim3((unsigned)(M1/64),Nc[2]/64),128,0,stream>>>(
        A2, wh[2], Bc[2], Gc[2], Ec[2], A3, (int)M1, Kp[2], Nc[2]);
    maxpool_kernel<<<grid1d((size_t)B*512*128,256),256,0,stream>>>(
        A3, l1pts, nullptr, B*512, 32, 128);
  }

  // ---- SA2: npoint=128, r=0.4, ns=64, input N=512
  fps_kernel<<<B,1024,0,stream>>>(nx1, 512, 128, fps2);
  gather3_kernel<<<grid1d((size_t)B*128,256),256,0,stream>>>(nx1, fps2, nx2, 512, 128, B);
  ballquery_kernel<<<B,512,(size_t)512*3*4,stream>>>(nx1, nx2, 512, 128, 64, 0.16f, idx2);
  {
    size_t M2 = (size_t)B*128*64;
    group_kernel<<<grid1d(M2,256),256,0,stream>>>(nx1, l1pts, nx2, idx2, Gb,
                                                  512, 128, 64, 128, 160, B);
    gemm_bn_relu_wmma<<<dim3((unsigned)(M2/64),Nc[3]/64),128,0,stream>>>(
        Gb, wh[3], Bc[3], Gc[3], Ec[3], A1, (int)M2, Kp[3], Nc[3]);
    gemm_bn_relu_wmma<<<dim3((unsigned)(M2/64),Nc[4]/64),128,0,stream>>>(
        A1, wh[4], Bc[4], Gc[4], Ec[4], A2, (int)M2, Kp[4], Nc[4]);
    gemm_bn_relu_wmma<<<dim3((unsigned)(M2/64),Nc[5]/64),128,0,stream>>>(
        A2, wh[5], Bc[5], Gc[5], Ec[5], A3, (int)M2, Kp[5], Nc[5]);
    maxpool_kernel<<<grid1d((size_t)B*128*256,256),256,0,stream>>>(
        A3, l2pts, nullptr, B*128, 64, 256);
  }

  // ---- SA3 (group-all over 128 points)
  {
    size_t M3 = (size_t)B*128;
    group_all_kernel<<<grid1d(M3,256),256,0,stream>>>(nx2, l2pts, Gb, 128, 256, 288, B);
    gemm_bn_relu_wmma<<<dim3((unsigned)(M3/64),Nc[6]/64),128,0,stream>>>(
        Gb, wh[6], Bc[6], Gc[6], Ec[6], A1, (int)M3, Kp[6], Nc[6]);
    gemm_bn_relu_wmma<<<dim3((unsigned)(M3/64),Nc[7]/64),128,0,stream>>>(
        A1, wh[7], Bc[7], Gc[7], Ec[7], A2, (int)M3, Kp[7], Nc[7]);
    gemm_bn_relu_wmma<<<dim3((unsigned)(M3/64),Nc[8]/64),128,0,stream>>>(
        A2, wh[8], Bc[8], Gc[8], Ec[8], A3, (int)M3, Kp[8], Nc[8]);
    maxpool_kernel<<<grid1d((size_t)B*1024,256),256,0,stream>>>(
        A3, l3, (float*)d_out + 1280, B, 128, 1024);
  }

  // ---- FC head
  fc_kernel<<<grid1d((size_t)B*512,256),256,0,stream>>>(l3,  fw1, fb1, fg1, fe1, fcb1, B, 1024, 512, 0);
  fc_kernel<<<grid1d((size_t)B*256,256),256,0,stream>>>(fcb1,fw2, fb2, fg2, fe2, fcb2, B, 512, 256, 0);
  fc_kernel<<<grid1d((size_t)B*40, 256),256,0,stream>>>(fcb2,fw3, fb3, nullptr, nullptr, fcb3, B, 256, 40, 1);
  logsoftmax_kernel<<<1,32,0,stream>>>(fcb3, (float*)d_out, B, 40);
}
